// Expert_13082470383822
// MI455X (gfx1250) — compile-verified, matrix operand order checked by reading, not run on hardware
//
#include <hip/hip_runtime.h>

// Problem constants (from reference): B=4, S=4096, D=2048, K_conv=4
constexpr int Bb   = 4;
constexpr int Ss   = 4096;
constexpr int Dd   = 2048;              // = K = N of the GEMM
constexpr int Mtot = Bb * Ss;           // 16384
constexpr int Ktot = Dd;                // 2048
constexpr int Ntot = Dd;                // 2048

// GEMM tiling: block 256(M) x 128(N), K-step 32, 8 wave32 as 4(M) x 2(N),
// wave tile 64x64 = 4x4 WMMA accumulators.
#define BM 256
#define BN 128
#define BK 32
#define LDS_STRIDE 40                    // halfwords/row (32 data + 8 pad)
#define ABUF_BYTES (BM * LDS_STRIDE * 2) // 20480 B per A buffer
#define BBUF_BYTES (BN * LDS_STRIDE * 2) // 10240 B per B buffer

typedef _Float16 v16h __attribute__((ext_vector_type(16)));
typedef _Float16 v8h  __attribute__((ext_vector_type(8)));
typedef float    v8f  __attribute__((ext_vector_type(8)));

// ---------------------------------------------------------------------------
// Stage 1: depthwise causal conv (K=4) + bias + SiLU, cast to f16
// ---------------------------------------------------------------------------
__global__ __launch_bounds__(256) void conv_silu_cast_f16(
    const float* __restrict__ x,       // [Mtot][Dd]
    const float* __restrict__ cw,      // [Dd][4]
    const float* __restrict__ cb,      // [Dd]
    _Float16* __restrict__ Yh)         // [Mtot][Dd] f16
{
    size_t idx = (size_t)blockIdx.x * 256 + threadIdx.x;   // m*Dd + d
    int d = (int)(idx & (Dd - 1));
    int m = (int)(idx >> 11);                               // Dd = 2^11
    int s = m & (Ss - 1);                                   // Ss = 2^12

    float w0 = cw[d * 4 + 0];
    float w1 = cw[d * 4 + 1];
    float w2 = cw[d * 4 + 2];
    float w3 = cw[d * 4 + 3];

    float v = x[idx] * w3 + cb[d];
    if (s >= 1) v += x[idx - (size_t)Dd]     * w2;
    if (s >= 2) v += x[idx - (size_t)2 * Dd] * w1;
    if (s >= 3) v += x[idx - (size_t)3 * Dd] * w0;

    v = v / (1.0f + __expf(-v));        // SiLU
    Yh[idx] = (_Float16)v;
}

__global__ __launch_bounds__(256) void cast_w_f16(
    const float* __restrict__ W, _Float16* __restrict__ Wh)
{
    size_t idx = (size_t)blockIdx.x * 256 + threadIdx.x;
    Wh[idx] = (_Float16)W[idx];
}

// ---------------------------------------------------------------------------
// Async global -> LDS copy (CDNA5): no VGPR staging, tracked by ASYNCcnt.
// GVS mode: 64-bit SGPR base + 32-bit per-lane byte offset.
// ---------------------------------------------------------------------------
__device__ __forceinline__ void async_copy_b128(uint32_t lds_addr,
                                                uint32_t byte_off,
                                                const void* base) {
    asm volatile("global_load_async_to_lds_b128 %0, %1, %2"
                 :
                 : "v"(lds_addr), "v"(byte_off), "s"(base)
                 : "memory");
}

// ---------------------------------------------------------------------------
// Stage 2: C[m,n] = sum_k A[m,k]*Bmat[n,k] + bias[n]   (f16 in, f32 accum)
// ---------------------------------------------------------------------------
__global__ __launch_bounds__(256) void gemm_wmma_f16(
    const _Float16* __restrict__ A,     // [Mtot][Ktot]
    const _Float16* __restrict__ Bm,    // [Ntot][Ktot]
    const float*    __restrict__ bias,  // [Ntot]
    float*          __restrict__ out)   // [Mtot][Ntot]
{
    __shared__ _Float16 As[2][BM * LDS_STRIDE];   // 40960 B
    __shared__ _Float16 Bs[2][BN * LDS_STRIDE];   // 20480 B  (total 60 KiB)

    const int tid   = threadIdx.x;
    const int wave  = tid >> 5;
    const int lane  = tid & 31;
    const int waveM = wave >> 1;       // 0..3  -> M offset waveM*64
    const int waveN = wave & 1;        // 0..1  -> N offset waveN*64
    const int mBase = blockIdx.y * BM;
    const int nBase = blockIdx.x * BN;

    const int l15     = lane & 15;
    const int halfSel = lane >> 4;     // 0 or 1
    const int khalf   = halfSel * 8;   // K sub-chunk select per ISA A-layout

    // LDS byte addresses (low 32 bits of generic pointer = LDS offset)
    const uint32_t asBase = (uint32_t)(uintptr_t)&As[0][0];
    const uint32_t bsBase = (uint32_t)(uintptr_t)&Bs[0][0];

    // Copy mapping: 16B chunks; chunk c = r*256 + tid -> row = c>>2 = r*64 + (tid>>2),
    // kc = tid&3 (constant per thread). A: r<4 (256 rows), B: r<2 (128 rows).
    const int t4 = tid >> 2;           // 0..63
    const int kc = tid & 3;            // 0..3
    const uint32_t aOff0 = (uint32_t)(((size_t)(mBase + t4) * Ktot + kc * 8) * 2);
    const uint32_t bOff0 = (uint32_t)(((size_t)(nBase + t4) * Ktot + kc * 8) * 2);
    const uint32_t ldsC0 = (uint32_t)((t4 * LDS_STRIDE + kc * 8) * 2);

    auto issue_stage = [&](int stage) {
        const uint32_t abuf = (uint32_t)(stage & 1) * ABUF_BYTES;
        const uint32_t bbuf = (uint32_t)(stage & 1) * BBUF_BYTES;
        const uint32_t k0b  = (uint32_t)(stage * BK * 2);   // byte offset in K
#pragma unroll
        for (int r = 0; r < 4; ++r)
            async_copy_b128(asBase + abuf + ldsC0 + r * (64 * LDS_STRIDE * 2),
                            aOff0 + k0b + r * (64 * Ktot * 2), A);
#pragma unroll
        for (int r = 0; r < 2; ++r)
            async_copy_b128(bsBase + bbuf + ldsC0 + r * (64 * LDS_STRIDE * 2),
                            bOff0 + k0b + r * (64 * Ktot * 2), Bm);
    };

    v8f acc[4][4];
#pragma unroll
    for (int mt = 0; mt < 4; ++mt)
#pragma unroll
        for (int nt = 0; nt < 4; ++nt)
#pragma unroll
            for (int e = 0; e < 8; ++e) acc[mt][nt][e] = 0.0f;

    issue_stage(0);

    const int nk = Ktot / BK;   // 64
    for (int kt = 0; kt < nk; ++kt) {
        const int buf = kt & 1;

        if (kt + 1 < nk) {
            // Buffer (kt+1)&1 was last read at iteration kt-1; trailing
            // __syncthreads() of kt-1 makes this write safe.
            issue_stage(kt + 1);
            // 6 ops/stage/thread, in-order completion: <=6 outstanding means
            // stage kt has fully landed in LDS.
            asm volatile("s_wait_asynccnt 0x6" ::: "memory");
        } else {
            asm volatile("s_wait_asynccnt 0x0" ::: "memory");
        }
        __syncthreads();   // all waves' stage-kt data visible

        // A fragments (ISA 16-bit 16x32 A layout): lane<16 row m, K{0..7,16..23};
        // lane>=16 same row, K{8..15,24..31}
        v16h af[4];
#pragma unroll
        for (int mt = 0; mt < 4; ++mt) {
            int row = waveM * 64 + mt * 16 + l15;
            const _Float16* p = &As[buf][row * LDS_STRIDE + khalf];
            v8h lo = *(const v8h*)p;
            v8h hi = *(const v8h*)(p + 16);
            af[mt] = __builtin_shufflevector(lo, hi, 0, 1, 2, 3, 4, 5, 6, 7,
                                             8, 9, 10, 11, 12, 13, 14, 15);
        }
        // B fragments: identical packing (lane holds column n)
        v16h bf[4];
#pragma unroll
        for (int nt = 0; nt < 4; ++nt) {
            int col = waveN * 64 + nt * 16 + l15;
            const _Float16* p = &Bs[buf][col * LDS_STRIDE + khalf];
            v8h lo = *(const v8h*)p;
            v8h hi = *(const v8h*)(p + 16);
            bf[nt] = __builtin_shufflevector(lo, hi, 0, 1, 2, 3, 4, 5, 6, 7,
                                             8, 9, 10, 11, 12, 13, 14, 15);
        }

#pragma unroll
        for (int mt = 0; mt < 4; ++mt)
#pragma unroll
            for (int nt = 0; nt < 4; ++nt)
                acc[mt][nt] = __builtin_amdgcn_wmma_f32_16x16x32_f16(
                    false, af[mt], false, bf[nt], (short)0, acc[mt][nt],
                    false, false);

        __syncthreads();   // all waves done reading buf before it is rewritten
    }

    // Epilogue: C/D layout — VGPR v: lanes 0-15 -> M=v, N=lane; lanes 16-31 -> M=8+v
#pragma unroll
    for (int nt = 0; nt < 4; ++nt) {
        int col = nBase + waveN * 64 + nt * 16 + l15;
        float bp = bias[col];
#pragma unroll
        for (int mt = 0; mt < 4; ++mt) {
            int row0 = mBase + waveM * 64 + mt * 16 + halfSel * 8;
#pragma unroll
            for (int v = 0; v < 8; ++v) {
                out[(size_t)(row0 + v) * Ntot + col] = acc[mt][nt][v] + bp;
            }
        }
    }
}

// ---------------------------------------------------------------------------
extern "C" void kernel_launch(void* const* d_in, const int* in_sizes, int n_in,
                              void* d_out, int out_size, void* d_ws, size_t ws_size,
                              hipStream_t stream) {
    const float* x  = (const float*)d_in[0];   // [B,S,D]
    const float* cw = (const float*)d_in[1];   // [D,4]
    const float* cb = (const float*)d_in[2];   // [D]
    const float* Wp = (const float*)d_in[3];   // [D,D]
    const float* bp = (const float*)d_in[4];   // [D]
    float* out = (float*)d_out;

    _Float16* Yh = (_Float16*)d_ws;                        // 64 MiB
    _Float16* Wh = Yh + (size_t)Mtot * Ktot;               // +8 MiB

    conv_silu_cast_f16<<<(int)(((size_t)Mtot * Dd) / 256), 256, 0, stream>>>(
        x, cw, cb, Yh);
    cast_w_f16<<<(int)(((size_t)Ntot * Ktot) / 256), 256, 0, stream>>>(Wp, Wh);

    dim3 grid(Ntot / BN, Mtot / BM);   // 16 x 64
    gemm_wmma_f16<<<grid, 256, 0, stream>>>(Yh, Wh, bp, out);
}
